// UpcasingTransformer_83202106458240
// MI455X (gfx1250) — matrix operand order repaired and checked by reading, not compiled, not run on hardware
//
#include <hip/hip_runtime.h>

// ---------------------------------------------------------------------------
// Problem constants (from the reference)
// ---------------------------------------------------------------------------
constexpr int kE = 1024;      // model dim
constexpr int kH = 16;        // heads
constexpr int kDH = 64;       // head dim
constexpr int kT = 2048;      // seq len
constexpr int kB = 4;         // batch
constexpr int kM = kB * kT;   // 8192 token rows
constexpr int kF = 4096;      // ffn dim
constexpr int kDepth = 8;
constexpr float kScale = 0.022097086912079608f;  // 1/sqrt(2048) (reference scales by seq len!)

// ---------------------------------------------------------------------------
// CDNA5 WMMA types / helpers (wave32, 16x16x32 bf16 -> f32)
// ---------------------------------------------------------------------------
typedef __attribute__((ext_vector_type(16))) __bf16 v16bf;
typedef __attribute__((ext_vector_type(8)))  __bf16 v8bf;
typedef __attribute__((ext_vector_type(8)))  float  v8f;

__device__ __forceinline__ v8f wmma_bf16(v16bf a, v16bf b, v8f c) {
  return __builtin_amdgcn_wmma_f32_16x16x32_bf16(false, a, false, b, (short)0, c,
                                                 false, false);
}

union AFrag { v16bf v; v8bf h[2]; };

__device__ __forceinline__ v8f vzero() {
  v8f z = {0.f, 0.f, 0.f, 0.f, 0.f, 0.f, 0.f, 0.f};
  return z;
}

// --- CDNA5 async global->LDS copy (ASYNCcnt-tracked) -----------------------
__device__ __forceinline__ void async_load_b128(unsigned lds_off, const void* gptr) {
  asm volatile("global_load_async_to_lds_b128 %0, %1, off"
               :: "v"(lds_off), "v"(gptr) : "memory");
}
__device__ __forceinline__ void wait_async0() {
  asm volatile("s_wait_asynccnt 0x0" ::: "memory");
}
__device__ __forceinline__ void wait_async1() {
  asm volatile("s_wait_asynccnt 0x1" ::: "memory");
}
// Wave-private LDS ordering: CDNA5 DS ops are in program order per wave; we
// only need to stop the compiler from reordering across the store->load edge.
__device__ __forceinline__ void wave_lds_fence() {
  asm volatile("" ::: "memory");
}

// ---------------------------------------------------------------------------
// Workspace layout (bytes). Weights are converted per-layer into a reused
// 24MB region so the live bf16 weight set stays L2-resident.
// ---------------------------------------------------------------------------
constexpr size_t OFF_X   = 0;                                    // f32 [M,E]
constexpr size_t OFF_H   = OFF_X   + (size_t)kM * kE * 4;        // bf16 [M,E]
constexpr size_t OFF_Q   = OFF_H   + (size_t)kM * kE * 2;        // bf16 [B*H,T,DH]
constexpr size_t OFF_K   = OFF_Q   + (size_t)kM * kE * 2;        // bf16 [B*H,T,DH]
constexpr size_t OFF_V   = OFF_K   + (size_t)kM * kE * 2;        // bf16 [B*H,DH,T] (transposed)
constexpr size_t OFF_ATT = OFF_V   + (size_t)kM * kE * 2;        // bf16 [M,E]
constexpr size_t OFF_U   = OFF_ATT + (size_t)kM * kE * 2;        // bf16 [M,F]
constexpr size_t OFF_WQ  = OFF_U   + (size_t)kM * kF * 2;        // bf16 [E,E] (Bt = [N,K])
constexpr size_t OFF_WK  = OFF_WQ  + (size_t)kE * kE * 2;
constexpr size_t OFF_WV  = OFF_WK  + (size_t)kE * kE * 2;
constexpr size_t OFF_WO  = OFF_WV  + (size_t)kE * kE * 2;
constexpr size_t OFF_W1  = OFF_WO  + (size_t)kE * kE * 2;        // bf16 [F,E]
constexpr size_t OFF_W2  = OFF_W1  + (size_t)kE * kF * 2;        // bf16 [E,F]

// ---------------------------------------------------------------------------
// Embedding: x[bt,e] = emb[tok[bt],e] + pos[t,e]
// ---------------------------------------------------------------------------
__global__ __launch_bounds__(256) void embed_kernel(
    const int* __restrict__ tok, const float* __restrict__ emb,
    const float* __restrict__ pos, float* __restrict__ x) {
  int gid = blockIdx.x * 256 + threadIdx.x;  // over M*E
  int bt = gid >> 10, e = gid & 1023;
  int t = bt & (kT - 1);
  x[gid] = emb[(size_t)tok[bt] * kE + e] + pos[(size_t)t * kE + e];
}

// ---------------------------------------------------------------------------
// LayerNorm over E=1024, one block per row, writes bf16 for WMMA consumption
// ---------------------------------------------------------------------------
__global__ __launch_bounds__(256) void ln_kernel(
    const float* __restrict__ x, const float* __restrict__ g,
    const float* __restrict__ b, __bf16* __restrict__ out) {
  __shared__ float red[16];
  int row = blockIdx.x;
  int tid = threadIdx.x;
  const float4 v = ((const float4*)(x + (size_t)row * kE))[tid];
  float s = v.x + v.y + v.z + v.w;
  float q = v.x * v.x + v.y * v.y + v.z * v.z + v.w * v.w;
  for (int m = 16; m >= 1; m >>= 1) {
    s += __shfl_xor(s, m, 32);
    q += __shfl_xor(q, m, 32);
  }
  int w = tid >> 5, lane = tid & 31;
  if (lane == 0) { red[w] = s; red[8 + w] = q; }
  __syncthreads();
  float ts = 0.f, tq = 0.f;
  for (int i = 0; i < 8; ++i) { ts += red[i]; tq += red[8 + i]; }
  float mu = ts * (1.f / kE);
  float rstd = rsqrtf(tq * (1.f / kE) - mu * mu + 1e-5f);
  const float4 gv = ((const float4*)g)[tid];
  const float4 bv = ((const float4*)b)[tid];
  __bf16* o = out + (size_t)row * kE + tid * 4;
  o[0] = (__bf16)((v.x - mu) * rstd * gv.x + bv.x);
  o[1] = (__bf16)((v.y - mu) * rstd * gv.y + bv.y);
  o[2] = (__bf16)((v.z - mu) * rstd * gv.z + bv.z);
  o[3] = (__bf16)((v.w - mu) * rstd * gv.w + bv.w);
}

// ---------------------------------------------------------------------------
// Weight conversion: fp32 -> bf16, laid out as Bt[N,K] (transposed) so that
// WMMA B-fragments become contiguous loads.
// ---------------------------------------------------------------------------
// Wq/Wk/Wv stored [H,E,DH]; Bt[n,k] with n = h*64+d, k = e.
__global__ __launch_bounds__(256) void qkv_cvt_kernel(
    const float* __restrict__ src, __bf16* __restrict__ dst) {
  int gid = blockIdx.x * 256 + threadIdx.x;  // over E*E
  int n = gid >> 10, e = gid & 1023;
  dst[gid] = (__bf16)src[((size_t)(n >> 6) << 16) + (e << 6) + (n & 63)];
}

// Generic transpose+convert: src[K,N] f32 -> dst[N,K] bf16
__global__ __launch_bounds__(256) void tr_cvt_kernel(
    const float* __restrict__ src, __bf16* __restrict__ dst, int K, int N) {
  int gid = blockIdx.x * 256 + threadIdx.x;  // over N*K
  int n = gid / K, k = gid - n * K;
  dst[gid] = (__bf16)src[(size_t)k * N + n];
}

// ---------------------------------------------------------------------------
// GEMM: C[M,N] = A[M,K](bf16, row-major) x Bt[N,K](bf16).
// Block = 256 threads = 8 waves; block tile 128x64; each wave owns 16 rows,
// all 64 cols (4 D-frags); K-step 32 -> 4 v_wmma per wave per step.
// The shared 64x32 B tile is staged through LDS with CDNA5 async
// global->LDS copies (double-buffered, ASYNCcnt-tracked), so the 8 waves
// read it from LDS instead of re-fetching it 8x from global.
// Rows are padded to 40 halves (80B) for conflict-free ds_load_b128.
// MODE 0: store bf16 into q/k layout [B*H,T,DH]
// MODE 1: store bf16 into v-transposed layout [B*H,DH,T]
// MODE 2: out = relu(C + bias) as bf16 [M,N]
// MODE 3: out(f32 [M,N]) += C (+bias)
// ---------------------------------------------------------------------------
template <int MODE>
__global__ __launch_bounds__(256) void gemm_kernel(
    const __bf16* __restrict__ A, const __bf16* __restrict__ Bt,
    const float* __restrict__ bias, void* __restrict__ out, int N, int K) {
  __shared__ __bf16 smemB[2][64 * 40];  // 2 x 5120B
  const int tid = threadIdx.x;
  const int lane = tid & 31;
  const int w = tid >> 5;
  const int g = lane >> 4;        // half-wave group
  const int ln = lane & 15;
  const int rowBase = blockIdx.y * 128 + w * 16;
  const int nBase = blockIdx.x * 64;

  // async-copy assignment: thread -> (row n 0..63, 16B chunk 0..3)
  const int cpN = tid >> 2;
  const int cpC = tid & 3;
  const __bf16* gsrc = Bt + (size_t)(nBase + cpN) * K + cpC * 8;
  const unsigned ldsRow = (unsigned)(uintptr_t)&smemB[0][cpN * 40 + cpC * 8];
  const unsigned ldsBufStride =
      (unsigned)((uintptr_t)&smemB[1][0] - (uintptr_t)&smemB[0][0]);

  v8f acc[4] = {vzero(), vzero(), vzero(), vzero()};
  const __bf16* arow = A + (size_t)(rowBase + ln) * K;

  // prologue: stage kk=0 into buffer 0
  async_load_b128(ldsRow, gsrc);

  for (int kk = 0; kk < K; kk += 32) {
    const int cur = (kk >> 5) & 1;
    const bool more = (kk + 32) < K;
    if (more)
      async_load_b128(ldsRow + (cur ^ 1) * ldsBufStride, gsrc + kk + 32);
    if (more) wait_async1(); else wait_async0();
    __syncthreads();  // all waves' chunks of the current tile are in LDS

    // A frag (16x32): lane ln holds row, half-wave g picks K chunks
    AFrag a;
    a.h[0] = *(const v8bf*)(arow + kk + g * 16);
    a.h[1] = *(const v8bf*)(arow + kk + g * 16 + 8);
#pragma unroll
    for (int j = 0; j < 4; ++j) {
      // B frag (32x16): lane ln = column n; contiguous K halves in LDS row
      AFrag bfr;
      const __bf16* bp = &smemB[cur][(j * 16 + ln) * 40 + g * 16];
      bfr.h[0] = *(const v8bf*)(bp);
      bfr.h[1] = *(const v8bf*)(bp + 8);
      acc[j] = wmma_bf16(a.v, bfr.v, acc[j]);
    }
    __syncthreads();  // everyone done reading before this buffer is re-filled
  }

#pragma unroll
  for (int j = 0; j < 4; ++j) {
#pragma unroll
    for (int r = 0; r < 8; ++r) {
      const int row = rowBase + r + 8 * g;   // D layout: M = r + 8*halfwave
      const int c = nBase + j * 16 + ln;     // D layout: N = lane&15 + subtile
      float val = acc[j][r];
      if constexpr (MODE == 0) {
        int b = row >> 11, t = row & (kT - 1), hh = c >> 6, d = c & 63;
        ((__bf16*)out)[(((size_t)(b * kH + hh)) * kT + t) * kDH + d] = (__bf16)val;
      } else if constexpr (MODE == 1) {
        int b = row >> 11, t = row & (kT - 1), hh = c >> 6, d = c & 63;
        ((__bf16*)out)[(((size_t)(b * kH + hh)) * kDH + d) * kT + t] = (__bf16)val;
      } else if constexpr (MODE == 2) {
        val += bias[c];
        ((__bf16*)out)[(size_t)row * N + c] = (__bf16)fmaxf(val, 0.f);
      } else {
        if (bias) val += bias[c];
        ((float*)out)[(size_t)row * N + c] += val;
      }
    }
  }
}

// ---------------------------------------------------------------------------
// Flash attention (non-causal, scale = 1/sqrt(T) per reference).
// One wave per (b,h, 16-query-row tile). Streams K/V in 32-col blocks:
//   S(16x32) = q(16x64) @ k^T                    -> 4 wmma
//   online max via half-wave shuffles, P = exp(S-m)
//   row-sums l via P @ ones(32x16)               -> 1 wmma (no shuffle tree)
//   O(16x64) += P(16x32) @ V                     -> 4 wmma
// P relayout D->A goes through a wave-private LDS tile; CDNA5 DS ops are
// in-order per wave, so only a compiler fence is needed (no block barrier).
// ---------------------------------------------------------------------------
__global__ __launch_bounds__(256) void flash_kernel(
    const __bf16* __restrict__ q, const __bf16* __restrict__ k,
    const __bf16* __restrict__ vT, __bf16* __restrict__ att) {
  __shared__ __bf16 pLds[8][16][32];  // per-wave 16x32 P tile
  const int lane = threadIdx.x & 31;
  const int w = threadIdx.x >> 5;
  const int g = lane >> 4;
  const int ln = lane & 15;
  const int widx = blockIdx.x * 8 + w;
  const int qt = widx & 127;   // query tile within (b,h)
  const int bh = widx >> 7;    // 0..63

  const __bf16* qb = q + (size_t)bh * kT * kDH;
  const __bf16* kb = k + (size_t)bh * kT * kDH;
  const __bf16* vb = vT + (size_t)bh * kDH * kT;

  // Load the 16x64 Q tile as two A-frags (K=0..31, K=32..63), kept in regs.
  AFrag qf[2];
  {
    const __bf16* qr = qb + (size_t)(qt * 16 + ln) * kDH;
    qf[0].h[0] = *(const v8bf*)(qr + g * 16);
    qf[0].h[1] = *(const v8bf*)(qr + g * 16 + 8);
    qf[1].h[0] = *(const v8bf*)(qr + 32 + g * 16);
    qf[1].h[1] = *(const v8bf*)(qr + 32 + g * 16 + 8);
  }

  v16bf ones;
#pragma unroll
  for (int i = 0; i < 16; ++i) ones[i] = (__bf16)1.0f;

  v8f o[4] = {vzero(), vzero(), vzero(), vzero()};
  v8f lacc = vzero();            // row sums, same D layout as o
  float mrow[8];
#pragma unroll
  for (int r = 0; r < 8; ++r) mrow[r] = -1e30f;

  for (int s0 = 0; s0 < kT; s0 += 32) {
    v8f sc[2] = {vzero(), vzero()};
#pragma unroll
    for (int nh = 0; nh < 2; ++nh) {
      const int col = s0 + nh * 16 + ln;  // key index = B column
#pragma unroll
      for (int kk = 0; kk < 2; ++kk) {
        v16bf bf = *(const v16bf*)(kb + (size_t)col * kDH + kk * 32 + g * 16);
        sc[nh] = wmma_bf16(qf[kk].v, bf, sc[nh]);
      }
    }

    // Online softmax per row r (row M = r + 8*g; all 32 cols in this half)
#pragma unroll
    for (int r = 0; r < 8; ++r) {
      float v0 = sc[0][r] * kScale;
      float v1 = sc[1][r] * kScale;
      float mx = fmaxf(v0, v1);
      for (int m = 8; m >= 1; m >>= 1) mx = fmaxf(mx, __shfl_xor(mx, m, 32));
      float newm = fmaxf(mrow[r], mx);
      float alpha = __expf(mrow[r] - newm);
      mrow[r] = newm;
      float p0 = __expf(v0 - newm);
      float p1 = __expf(v1 - newm);
      lacc[r] *= alpha;
#pragma unroll
      for (int j = 0; j < 4; ++j) o[j][r] *= alpha;
      pLds[w][r + 8 * g][ln] = (__bf16)p0;
      pLds[w][r + 8 * g][ln + 16] = (__bf16)p1;
    }
    wave_lds_fence();  // DS in-order per wave; just stop compiler reordering

    // Re-read P as an A-frag (16x32)
    AFrag pf;
    const __bf16* pr = &pLds[w][ln][0];
    pf.h[0] = *(const v8bf*)(pr + g * 16);
    pf.h[1] = *(const v8bf*)(pr + g * 16 + 8);
    wave_lds_fence();

    // Row sums via WMMA against ones: no shuffle tree needed
    lacc = wmma_bf16(pf.v, ones, lacc);

    // O += P @ V  (V transposed: row d contiguous in s)
#pragma unroll
    for (int j = 0; j < 4; ++j) {
      v16bf bf = *(const v16bf*)(vb + (size_t)(j * 16 + ln) * kT + s0 + g * 16);
      o[j] = wmma_bf16(pf.v, bf, o[j]);
    }
  }

  const int b = bh >> 4, hh = bh & 15;
#pragma unroll
  for (int r = 0; r < 8; ++r) {
    float inv = 1.f / lacc[r];
    int bt = b * kT + qt * 16 + r + 8 * g;
#pragma unroll
    for (int j = 0; j < 4; ++j) {
      att[(size_t)bt * kE + hh * kDH + j * 16 + ln] = (__bf16)(o[j][r] * inv);
    }
  }
}

// ---------------------------------------------------------------------------
// Head: out[bt,0:2] = h[bt,:] @ Wu + bu   (one wave per token)
// ---------------------------------------------------------------------------
__global__ __launch_bounds__(256) void head_kernel(
    const __bf16* __restrict__ h, const float* __restrict__ Wu,
    const float* __restrict__ bu, float* __restrict__ out) {
  int lane = threadIdx.x & 31, w = threadIdx.x >> 5;
  int bt = blockIdx.x * 8 + w;
  const __bf16* hr = h + (size_t)bt * kE;
  float a0 = 0.f, a1 = 0.f;
  for (int e = lane; e < kE; e += 32) {
    float hv = (float)hr[e];
    a0 += hv * Wu[2 * e];
    a1 += hv * Wu[2 * e + 1];
  }
  for (int m = 16; m >= 1; m >>= 1) {
    a0 += __shfl_xor(a0, m, 32);
    a1 += __shfl_xor(a1, m, 32);
  }
  if (lane == 0) {
    out[(size_t)bt * 2 + 0] = a0 + bu[0];
    out[(size_t)bt * 2 + 1] = a1 + bu[1];
  }
}

// ---------------------------------------------------------------------------
// Host orchestration
// ---------------------------------------------------------------------------
extern "C" void kernel_launch(void* const* d_in, const int* in_sizes, int n_in,
                              void* d_out, int out_size, void* d_ws, size_t ws_size,
                              hipStream_t stream) {
  (void)in_sizes; (void)n_in; (void)out_size; (void)ws_size;
  const int*   tokens = (const int*)d_in[0];
  const float* emb_w  = (const float*)d_in[1];
  const float* pos_w  = (const float*)d_in[2];
  const float* Wq     = (const float*)d_in[3];
  const float* Wk     = (const float*)d_in[4];
  const float* Wv     = (const float*)d_in[5];
  const float* Wo     = (const float*)d_in[6];
  const float* ln1_g  = (const float*)d_in[7];
  const float* ln1_b  = (const float*)d_in[8];
  const float* W1     = (const float*)d_in[9];
  const float* b1     = (const float*)d_in[10];
  const float* W2     = (const float*)d_in[11];
  const float* b2     = (const float*)d_in[12];
  const float* ln2_g  = (const float*)d_in[13];
  const float* ln2_b  = (const float*)d_in[14];
  const float* lnf_g  = (const float*)d_in[15];
  const float* lnf_b  = (const float*)d_in[16];
  const float* Wu     = (const float*)d_in[17];
  const float* bu     = (const float*)d_in[18];

  char* ws = (char*)d_ws;
  float*  x    = (float*)(ws + OFF_X);
  __bf16* hb   = (__bf16*)(ws + OFF_H);
  __bf16* qb   = (__bf16*)(ws + OFF_Q);
  __bf16* kb   = (__bf16*)(ws + OFF_K);
  __bf16* vb   = (__bf16*)(ws + OFF_V);
  __bf16* attb = (__bf16*)(ws + OFF_ATT);
  __bf16* ub   = (__bf16*)(ws + OFF_U);
  __bf16* wqT  = (__bf16*)(ws + OFF_WQ);
  __bf16* wkT  = (__bf16*)(ws + OFF_WK);
  __bf16* wvT  = (__bf16*)(ws + OFF_WV);
  __bf16* woT  = (__bf16*)(ws + OFF_WO);
  __bf16* w1T  = (__bf16*)(ws + OFF_W1);
  __bf16* w2T  = (__bf16*)(ws + OFF_W2);

  const dim3 gE(kE / 64, kM / 128);   // N=1024 GEMMs
  const dim3 gF(kF / 64, kM / 128);   // N=4096 GEMM

  embed_kernel<<<kM * kE / 256, 256, 0, stream>>>(tokens, emb_w, pos_w, x);

  for (int l = 0; l < kDepth; ++l) {
    const size_t wEE = (size_t)l * kE * kE;
    const size_t wEF = (size_t)l * kE * kF;

    ln_kernel<<<kM, 256, 0, stream>>>(x, ln1_g + l * kE, ln1_b + l * kE, hb);

    qkv_cvt_kernel<<<kE * kE / 256, 256, 0, stream>>>(Wq + wEE, wqT);
    qkv_cvt_kernel<<<kE * kE / 256, 256, 0, stream>>>(Wk + wEE, wkT);
    qkv_cvt_kernel<<<kE * kE / 256, 256, 0, stream>>>(Wv + wEE, wvT);

    gemm_kernel<0><<<gE, 256, 0, stream>>>(hb, wqT, nullptr, qb, kE, kE);
    gemm_kernel<0><<<gE, 256, 0, stream>>>(hb, wkT, nullptr, kb, kE, kE);
    gemm_kernel<1><<<gE, 256, 0, stream>>>(hb, wvT, nullptr, vb, kE, kE);

    flash_kernel<<<kB * kH * (kT / 16) / 8, 256, 0, stream>>>(qb, kb, vb, attb);

    tr_cvt_kernel<<<kE * kE / 256, 256, 0, stream>>>(Wo + wEE, woT, kE, kE);
    gemm_kernel<3><<<gE, 256, 0, stream>>>(attb, woT, nullptr, x, kE, kE);

    ln_kernel<<<kM, 256, 0, stream>>>(x, ln2_g + l * kE, ln2_b + l * kE, hb);

    tr_cvt_kernel<<<kE * kF / 256, 256, 0, stream>>>(W1 + wEF, w1T, kE, kF);
    gemm_kernel<2><<<gF, 256, 0, stream>>>(hb, w1T, b1 + (size_t)l * kF, ub, kF, kE);

    tr_cvt_kernel<<<kE * kF / 256, 256, 0, stream>>>(W2 + wEF, w2T, kF, kE);
    gemm_kernel<3><<<gE, 256, 0, stream>>>(ub, w2T, b2 + (size_t)l * kE, x, kE, kF);
  }

  ln_kernel<<<kM, 256, 0, stream>>>(x, lnf_g, lnf_b, hb);
  head_kernel<<<kM / 8, 256, 0, stream>>>(hb, Wu, bu, (float*)d_out);
}
